// SchNetModel_10282151706848
// MI455X (gfx1250) — matrix operand order repaired and checked by reading, not compiled, not run on hardware
//
#include <hip/hip_runtime.h>
#include <hip/hip_bf16.h>
#include <stdint.h>

// ---------------- problem constants ----------------
#define NA   20000      // atoms
#define NE   640000     // edges
#define NG_  1000       // graphs
#define HC   128        // hidden channels
#define GNG  50         // num gaussians
#define GP   64         // gaussians padded to 64 for K-tiling
#define NL   6          // interaction layers
#define CUT  10.0f

typedef __attribute__((ext_vector_type(16))) __bf16 bf16x16;
typedef __attribute__((ext_vector_type(8)))  float  f32x8;

union FragBF {
    bf16x16 v;
    unsigned short h[16];
};

__device__ __forceinline__ unsigned short f2bf(float f) {
    unsigned u = __float_as_uint(f);
    unsigned r = u + 0x7FFFu + ((u >> 16) & 1u);   // round-to-nearest-even
    return (unsigned short)(r >> 16);
}

__device__ __forceinline__ float sspf(float x) {
    // shifted softplus: softplus(x) - log(2), numerically stable
    float ax = fabsf(x);
    float sp = fmaxf(x, 0.0f) + __logf(1.0f + __expf(-ax));
    return sp - 0.69314718055994531f;
}

// =====================================================================
// Kernel: embedding gather  h[i,:] = emb[z[i],:]
// =====================================================================
__global__ void embed_kernel(const int* __restrict__ z,
                             const float* __restrict__ emb,
                             float* __restrict__ h) {
    size_t t = (size_t)blockIdx.x * blockDim.x + threadIdx.x;
    if (t < (size_t)NA * HC) {
        int i = (int)(t >> 7);
        int j = (int)(t & 127);
        h[t] = emb[(size_t)z[i] * HC + j];
    }
}

// =====================================================================
// Kernel: edge geometry -> cosine cutoff + gaussian smearing (bf16, padded)
// =====================================================================
__global__ void edge_geom_kernel(const float* __restrict__ pos,
                                 const int* __restrict__ eidx,   // [2,E]
                                 unsigned short* __restrict__ eabf, // [E,GP] bf16
                                 float* __restrict__ Cbuf) {
    int e = blockIdx.x * blockDim.x + threadIdx.x;
    if (e >= NE) return;
    int r = eidx[e];
    int c = eidx[NE + e];
    float dx = pos[r * 3 + 0] - pos[c * 3 + 0];
    float dy = pos[r * 3 + 1] - pos[c * 3 + 1];
    float dz = pos[r * 3 + 2] - pos[c * 3 + 2];
    float d = sqrtf(dx * dx + dy * dy + dz * dz);
    Cbuf[e] = 0.5f * (__cosf(d * (3.14159265358979323846f / CUT)) + 1.0f);
    const float delta = CUT / (float)(GNG - 1);
    const float coeff = -0.5f / (delta * delta);
    unsigned short* o = eabf + (size_t)e * GP;
#pragma unroll 8
    for (int g = 0; g < GP; ++g) {
        float off = (float)g * delta;
        float t = d - off;
        float v = (g < GNG) ? __expf(coeff * t * t) : 0.0f;
        o[g] = f2bf(v);
    }
}

// =====================================================================
// Generic tiled GEMM: out[M,NCOLS] = epilogue(A[M,128] @ Bw[128,NCOLS] + bias)
// bf16 WMMA 16x16x32, f32 accumulate. 256 threads = 8 waves,
// block tile 128 rows; each wave owns a 16-row stripe x all NCOLS cols.
// =====================================================================
template <int NCOLS, bool ACT_SSP, bool RESID>
__global__ void gemm_bf16_kernel(const float* __restrict__ A,
                                 const float* __restrict__ Bw,
                                 const float* __restrict__ bias,
                                 const float* __restrict__ resid,
                                 float* __restrict__ out,
                                 int M) {
    extern __shared__ unsigned short smem[];
    unsigned short* Alds = smem;                    // 128*128 bf16 (row major)
    unsigned short* Blds = smem + 128 * 128;        // 128*NCOLS bf16 (frag-swizzled)
    float* biasLds = (float*)(smem + 128 * 128 + 128 * NCOLS);

    const int tid  = threadIdx.x;
    const int lane = tid & 31;
    const int wave = tid >> 5;
    const int m0   = blockIdx.x * 128;
    constexpr int NT = NCOLS / 16;

    // ---- fill A tile: f32 global -> bf16 LDS, zero-pad rows >= M ----
    for (int idx = tid; idx < 128 * 32; idx += 256) {
        int r  = idx >> 5;
        int c4 = (idx & 31) << 2;
        int gm = m0 + r;
        float4 f = make_float4(0.f, 0.f, 0.f, 0.f);
        if (gm < M) f = *(const float4*)(A + (size_t)gm * 128 + c4);
        unsigned short* dst = Alds + r * 128 + c4;
        dst[0] = f2bf(f.x); dst[1] = f2bf(f.y); dst[2] = f2bf(f.z); dst[3] = f2bf(f.w);
    }
    // ---- fill B tile into per-lane-contiguous fragment layout ----
    for (int e = tid; e < 128 * NCOLS; e += 256) {
        int kv = e / NCOLS;
        int n  = e - kv * NCOLS;
        int kt = kv >> 5, rem = kv & 31, hi = (rem >> 3) & 1, i = (rem & 7) + ((rem >> 4) << 3);
        int ln = hi * 16 + (n & 15), nt = n >> 4;
        Blds[((kt * NT + nt) * 32 + ln) * 16 + i] = f2bf(Bw[(size_t)kv * NCOLS + n]);
    }
    if (tid < NCOLS) biasLds[tid] = bias[tid];
    __syncthreads();

    // speculative prefetch of next block's A tile (global_prefetch_b8)
    if (tid < 16 && (m0 + 128) < M)
        __builtin_prefetch((const char*)(A + (size_t)(m0 + 128) * 128) + tid * 4096, 0, 0);

    f32x8 acc[NT];
#pragma unroll
    for (int t = 0; t < NT; ++t) acc[t] = (f32x8){0, 0, 0, 0, 0, 0, 0, 0};

    const int mrow = (wave << 4) + (lane & 15);
    const int half = lane >> 4;

#pragma unroll
    for (int kt = 0; kt < 4; ++kt) {
        const int kk = kt * 32;
        FragBF a;
        const unsigned short* ap = Alds + mrow * 128 + kk + half * 8;
#pragma unroll
        for (int i = 0; i < 8; ++i) a.h[i] = ap[i];
#pragma unroll
        for (int i = 0; i < 8; ++i) a.h[8 + i] = ap[16 + i];
#pragma unroll
        for (int nt = 0; nt < NT; ++nt) {
            FragBF b;
            const unsigned short* bp = Blds + ((kt * NT + nt) * 32 + lane) * 16;
#pragma unroll
            for (int i = 0; i < 16; ++i) b.h[i] = bp[i];
            acc[nt] = __builtin_amdgcn_wmma_f32_16x16x32_bf16(
                false, a.v, false, b.v, (short)0, acc[nt], false, false);
        }
    }

    // ---- epilogue: C/D layout -> lanes 0-15: M=r, lanes 16-31: M=8+r ----
    const int rbase = (wave << 4) + half * 8;
#pragma unroll
    for (int nt = 0; nt < NT; ++nt) {
        const int n = nt * 16 + (lane & 15);
#pragma unroll
        for (int r = 0; r < 8; ++r) {
            int gm = m0 + rbase + r;
            if (gm < M) {
                float v = acc[nt][r] + biasLds[n];
                if (ACT_SSP) v = sspf(v);
                if (RESID)   v += resid[(size_t)gm * NCOLS + n];
                out[(size_t)gm * NCOLS + n] = v;
            }
        }
    }
}

// =====================================================================
// Fused edge-filter kernel (per layer, 128-edge tiles):
//   t = ssp(ea @ w1 + b1)          (WMMA stage 1, K=64)
//   W = ssp(t  @ w2 + b2) * C      (WMMA stage 2, K=128)
//   conv[col[e],:] += hw[row[e],:] * W[e,:]   (gather + atomic scatter)
// ea tile is staged HBM -> LDS with the gfx1250 async-to-LDS engine.
// =====================================================================
__global__ void edge_filter_kernel(const unsigned short* __restrict__ eabf,
                                   const float* __restrict__ Cbuf,
                                   const int* __restrict__ erow,
                                   const int* __restrict__ ecol,
                                   const float* __restrict__ w1,
                                   const float* __restrict__ b1,
                                   const float* __restrict__ w2,
                                   const float* __restrict__ b2,
                                   const float* __restrict__ hw,
                                   float* __restrict__ conv) {
    extern __shared__ unsigned short smem[];
    unsigned short* eaA = smem;                 // 128*64 bf16 (stage1 A) \ reused
    unsigned short* B1  = smem + 8192;          // 64*128 bf16 frags      /  as B2
    unsigned short* B2  = smem;                 // 128*128 bf16 frags (after stage1)
    unsigned short* tA  = smem + 16384;         // 128*128 bf16 (stage2 A)
    float* fp   = (float*)(smem + 32768);
    float* bi1  = fp;
    float* bi2  = fp + 128;
    float* cv   = fp + 256;
    int*   rows = (int*)(fp + 384);
    int*   cols = rows + 128;

    const int tid  = threadIdx.x;
    const int lane = tid & 31;
    const int wave = tid >> 5;
    const int eb   = blockIdx.x * 128;

    // ---- ea tile (16 KB contiguous) : async HBM -> LDS, no VGPR staging ----
    {
        unsigned ldsoff = (unsigned)(uintptr_t)eaA + (unsigned)tid * 16u;
        unsigned long long ga =
            (unsigned long long)(uintptr_t)(eabf + (size_t)eb * GP) + (unsigned long long)tid * 16ull;
        asm volatile("global_load_async_to_lds_b128 %0, %1, off\n\t"
                     "global_load_async_to_lds_b128 %0, %1, off offset:4096\n\t"
                     "global_load_async_to_lds_b128 %0, %1, off offset:8192\n\t"
                     "global_load_async_to_lds_b128 %0, %1, off offset:12288"
                     :: "v"(ldsoff), "v"(ga) : "memory");
    }
    // ---- fill B1 fragments (K padded 50->64 with zeros) ----
    for (int e = tid; e < GP * 128; e += 256) {
        int kv = e >> 7, n = e & 127;
        int kt = kv >> 5, rem = kv & 31, hi = (rem >> 3) & 1, i = (rem & 7) + ((rem >> 4) << 3);
        int ln = hi * 16 + (n & 15), nt = n >> 4;
        float v = (kv < GNG) ? w1[kv * 128 + n] : 0.0f;
        B1[((kt * 8 + nt) * 32 + ln) * 16 + i] = f2bf(v);
    }
    if (tid < 128) {
        bi1[tid] = b1[tid];
        bi2[tid] = b2[tid];
        int e = eb + tid;
        rows[tid] = erow[e];
        cols[tid] = ecol[e];
        cv[tid]   = Cbuf[e];
    }
    asm volatile("s_wait_asynccnt 0" ::: "memory");
    __syncthreads();

    const int mrow = (wave << 4) + (lane & 15);
    const int half = lane >> 4;
    const int rbase = (wave << 4) + half * 8;

    // ---------------- stage 1: t = ssp(ea @ w1 + b1) ----------------
    {
        f32x8 acc[8];
#pragma unroll
        for (int t = 0; t < 8; ++t) acc[t] = (f32x8){0, 0, 0, 0, 0, 0, 0, 0};
#pragma unroll
        for (int kt = 0; kt < 2; ++kt) {
            FragBF a;
            const unsigned short* ap = eaA + mrow * GP + kt * 32 + half * 8;
#pragma unroll
            for (int i = 0; i < 8; ++i) a.h[i] = ap[i];
#pragma unroll
            for (int i = 0; i < 8; ++i) a.h[8 + i] = ap[16 + i];
#pragma unroll
            for (int nt = 0; nt < 8; ++nt) {
                FragBF b;
                const unsigned short* bp = B1 + ((kt * 8 + nt) * 32 + lane) * 16;
#pragma unroll
                for (int i = 0; i < 16; ++i) b.h[i] = bp[i];
                acc[nt] = __builtin_amdgcn_wmma_f32_16x16x32_bf16(
                    false, a.v, false, b.v, (short)0, acc[nt], false, false);
            }
        }
#pragma unroll
        for (int nt = 0; nt < 8; ++nt) {
            const int n = nt * 16 + (lane & 15);
#pragma unroll
            for (int r = 0; r < 8; ++r)
                tA[(rbase + r) * 128 + n] = f2bf(sspf(acc[nt][r] + bi1[n]));
        }
    }
    __syncthreads();

    // ---- fill B2 fragments (reuses eaA/B1 space) ----
    for (int e = tid; e < 128 * 128; e += 256) {
        int kv = e >> 7, n = e & 127;
        int kt = kv >> 5, rem = kv & 31, hi = (rem >> 3) & 1, i = (rem & 7) + ((rem >> 4) << 3);
        int ln = hi * 16 + (n & 15), nt = n >> 4;
        B2[((kt * 8 + nt) * 32 + ln) * 16 + i] = f2bf(w2[kv * 128 + n]);
    }
    __syncthreads();

    // ------- stage 2: W = ssp(t @ w2 + b2) * C; fused msg + scatter -------
    {
        f32x8 acc[8];
#pragma unroll
        for (int t = 0; t < 8; ++t) acc[t] = (f32x8){0, 0, 0, 0, 0, 0, 0, 0};
#pragma unroll
        for (int kt = 0; kt < 4; ++kt) {
            FragBF a;
            const unsigned short* ap = tA + mrow * 128 + kt * 32 + half * 8;
#pragma unroll
            for (int i = 0; i < 8; ++i) a.h[i] = ap[i];
#pragma unroll
            for (int i = 0; i < 8; ++i) a.h[8 + i] = ap[16 + i];
#pragma unroll
            for (int nt = 0; nt < 8; ++nt) {
                FragBF b;
                const unsigned short* bp = B2 + ((kt * 8 + nt) * 32 + lane) * 16;
#pragma unroll
                for (int i = 0; i < 16; ++i) b.h[i] = bp[i];
                acc[nt] = __builtin_amdgcn_wmma_f32_16x16x32_bf16(
                    false, a.v, false, b.v, (short)0, acc[nt], false, false);
            }
        }
#pragma unroll
        for (int nt = 0; nt < 8; ++nt) {
            const int n = nt * 16 + (lane & 15);
#pragma unroll
            for (int r = 0; r < 8; ++r) {
                const int ml = rbase + r;                       // edge within tile
                float Wv = sspf(acc[nt][r] + bi2[n]) * cv[ml];
                float hv = hw[(size_t)rows[ml] * 128 + n];      // gather
                atomicAdd(conv + (size_t)cols[ml] * 128 + n, hv * Wv); // scatter
            }
        }
    }
}

// =====================================================================
// Head stage 2: y[i] = t2[i,:] @ head_w2 + b2 ; out[batch[i]] += y[i]
// =====================================================================
__global__ void head2_kernel(const float* __restrict__ t2,
                             const float* __restrict__ w2,
                             const float* __restrict__ b2,
                             const int* __restrict__ batch,
                             float* __restrict__ out) {
    int i = blockIdx.x * blockDim.x + threadIdx.x;
    if (i >= NA) return;
    float a = b2[0];
#pragma unroll 8
    for (int j = 0; j < 64; ++j) a += t2[(size_t)i * 64 + j] * w2[j];
    atomicAdd(out + batch[i], a);
}

// =====================================================================
// host-side launch
// =====================================================================
extern "C" void kernel_launch(void* const* d_in, const int* in_sizes, int n_in,
                              void* d_out, int out_size, void* d_ws, size_t ws_size,
                              hipStream_t stream) {
    (void)in_sizes; (void)n_in; (void)ws_size;
    const float* pos     = (const float*)d_in[0];
    const int*   z       = (const int*)d_in[1];
    const int*   batch   = (const int*)d_in[2];
    const int*   eidx    = (const int*)d_in[3];
    const float* emb     = (const float*)d_in[4];
    const float* mlp_w1  = (const float*)d_in[5];
    const float* mlp_b1  = (const float*)d_in[6];
    const float* mlp_w2  = (const float*)d_in[7];
    const float* mlp_b2  = (const float*)d_in[8];
    const float* aw_w    = (const float*)d_in[9];
    const float* aw_b    = (const float*)d_in[10];
    const float* out_w1  = (const float*)d_in[11];
    const float* out_b1  = (const float*)d_in[12];
    const float* out_w2  = (const float*)d_in[13];
    const float* out_b2  = (const float*)d_in[14];
    const float* head_w1 = (const float*)d_in[15];
    const float* head_b1 = (const float*)d_in[16];
    const float* head_w2 = (const float*)d_in[17];
    const float* head_b2 = (const float*)d_in[18];
    float*       outF    = (float*)d_out;

    // workspace carve-out
    char* ws = (char*)d_ws;
    float* h    = (float*)(ws);                         // N*H f32
    float* hw   = (float*)(ws + 10240000);              // N*H f32
    float* conv = (float*)(ws + 20480000);              // N*H f32
    float* t1   = (float*)(ws + 30720000);              // N*H f32
    float* t2   = (float*)(ws + 40960000);              // N*64 f32
    float* Cbuf = (float*)(ws + 46080000);              // E f32
    unsigned short* eabf = (unsigned short*)(ws + 48640000); // E*64 bf16

    const size_t ldsGemm128 = (size_t)(128 * 128 + 128 * 128) * 2 + 128 * 4;
    const size_t ldsGemm64  = (size_t)(128 * 128 + 128 * 64) * 2 + 64 * 4;
    const size_t ldsEdge    = 65536 + 640 * 4;
    const int gemmGrid = (NA + 127) / 128;

    (void)hipMemsetAsync(d_out, 0, (size_t)out_size * sizeof(float), stream);

    embed_kernel<<<(NA * HC + 255) / 256, 256, 0, stream>>>(z, emb, h);
    edge_geom_kernel<<<(NE + 255) / 256, 256, 0, stream>>>(pos, eidx, eabf, Cbuf);

    for (int l = 0; l < NL; ++l) {
        const size_t o2 = (size_t)l * HC * HC;
        const size_t o1 = (size_t)l * GNG * HC;
        const size_t ob = (size_t)l * HC;
        // hw = h @ aw_w + aw_b
        gemm_bf16_kernel<128, false, false><<<gemmGrid, 256, ldsGemm128, stream>>>(
            h, aw_w + o2, aw_b + ob, nullptr, hw, NA);
        (void)hipMemsetAsync(conv, 0, (size_t)NA * HC * sizeof(float), stream);
        // fused edge filter + message + scatter
        edge_filter_kernel<<<NE / 128, 256, ldsEdge, stream>>>(
            eabf, Cbuf, eidx, eidx + NE,
            mlp_w1 + o1, mlp_b1 + ob, mlp_w2 + o2, mlp_b2 + ob, hw, conv);
        // t1 = ssp(conv @ out_w1 + b1)
        gemm_bf16_kernel<128, true, false><<<gemmGrid, 256, ldsGemm128, stream>>>(
            conv, out_w1 + o2, out_b1 + ob, nullptr, t1, NA);
        // h = h + t1 @ out_w2 + b2
        gemm_bf16_kernel<128, false, true><<<gemmGrid, 256, ldsGemm128, stream>>>(
            t1, out_w2 + o2, out_b2 + ob, h, h, NA);
    }
    // head
    gemm_bf16_kernel<64, true, false><<<gemmGrid, 256, ldsGemm64, stream>>>(
        h, head_w1, head_b1, nullptr, t2, NA);
    head2_kernel<<<(NA + 255) / 256, 256, 0, stream>>>(t2, head_w2, head_b2, batch, outF);
}
